// GIPAConv_65970697666604
// MI455X (gfx1250) — matrix operand order repaired and checked by reading, not compiled, not run on hardware
//
#include <hip/hip_runtime.h>
#include <math.h>

typedef _Float16 v8h  __attribute__((ext_vector_type(8)));
typedef _Float16 v16h __attribute__((ext_vector_type(16)));
typedef float    v8f  __attribute__((ext_vector_type(8)));

#define F_NODE 128
#define H_ATT  64
#define HEADS  8
#define PDIM   16

// ---------------- utility kernels ----------------

__global__ void k_f32_to_f16(const float* __restrict__ src, _Float16* __restrict__ dst, int n) {
    int i = blockIdx.x * blockDim.x + threadIdx.x;
    if (i < n) dst[i] = (_Float16)src[i];
}

__global__ void k_zero_u32(unsigned int* __restrict__ p, int n) {
    int i = blockIdx.x * blockDim.x + threadIdx.x;
    if (i < n) p[i] = 0u;
}

// Swizzle a row-major f32 weight [Ktot, Ntot] into per-fragment B layout:
// out[((nt*ksteps + ks)*32 + lane)*16 + i] = W[K][N]
//   N = nt*16 + (lane & 15)
//   K = ks*32 + 16*(lane >= 16) + i          (contiguous k order, same as the A loads)
__global__ void k_swizzle_b(const float* __restrict__ W, _Float16* __restrict__ dst,
                            int Ktot, int Ntot, int ksteps, int total) {
    int tid = blockIdx.x * blockDim.x + threadIdx.x;
    if (tid >= total) return;
    int i    = tid & 15;
    int lane = (tid >> 4) & 31;
    int ks   = (tid >> 9) % ksteps;
    int nt   = (tid >> 9) / ksteps;
    int K = ks * 32 + ((lane >> 4) << 4) + i;
    int N = nt * 16 + (lane & 15);
    float v = (K < Ktot && N < Ntot) ? W[K * Ntot + N] : 0.0f;
    dst[tid] = (_Float16)v;
}

// ---------------- fused node GEMM kernel (WMMA) ----------------
// Each wave handles a 16-node row tile. Computes:
//   prop_src = feat @ Wps + bps            -> ws
//   out      = feat @ Wpd + bpd            -> d_out (residual base, atomics add later)
//   a_src    = relu(feat @ Was0) @ was1    -> ws
//   a_dst    = relu(feat @ Wad0) @ wad1    -> ws
__global__ void __launch_bounds__(256)
k_node_fused(const _Float16* __restrict__ feat16,
             const _Float16* __restrict__ wps_sw,   // [8][4][32][16]
             const _Float16* __restrict__ wpd_sw,   // [8][4][32][16]
             const _Float16* __restrict__ was0_sw,  // [4][4][32][16]
             const _Float16* __restrict__ wad0_sw,  // [4][4][32][16]
             const float* __restrict__ w_att_src1,  // [64][8] f32
             const float* __restrict__ w_att_dst1,  // [64][8] f32
             const float* __restrict__ b_prop_src,  // [128]
             const float* __restrict__ b_prop_dst,  // [128]
             float* __restrict__ prop_src,          // [N][128]
             float* __restrict__ a_src,             // [N][8]
             float* __restrict__ a_dst,             // [N][8]
             float* __restrict__ out,               // [N][128]
             int n_tiles, int n_nodes) {
    __shared__ float hid[8][16 * H_ATT];   // per-wave hidden scratch (32 KB)

    const int wave = threadIdx.x >> 5;
    const int lane = threadIdx.x & 31;
    int tile = blockIdx.x * 8 + wave;
    const bool valid = tile < n_tiles;
    const int t = valid ? tile : (n_tiles - 1);

    const int col   = lane & 15;          // C/D column, also A row within tile
    const int mbase = (lane >> 4) << 3;   // C/D row offset: lanes>=16 hold M=8..15
    const int koff  = (lane >> 4) << 3;   // unused; A loads give lanes>=16 K=16..31 (contiguous)

    // full tile => branch-free store path (always true when N % 16 == 0)
    const bool full = valid && ((t + 1) * 16 <= n_nodes);

    // ---- load A fragments (16x32 f16, 4 K-steps) ----
    int row = t * 16 + col;
    if (row >= n_nodes) row = n_nodes - 1;
    const _Float16* fp = feat16 + (size_t)row * F_NODE;
    // prefetch next tile's rows (streaming): lowers to global_prefetch_b8
    __builtin_prefetch(fp + 16 * F_NODE, 0, 0);
    v16h a[4];
#pragma unroll
    for (int ks = 0; ks < 4; ++ks) {
        a[ks] = *(const v16h*)(fp + ks * 32 + ((lane >> 4) << 4));
    }

    // ---- prop_src and prop_dst (-> out): 8 N-tiles x 4 K-steps each ----
    {
        float* ps_base = prop_src + (size_t)(t * 16 + mbase) * F_NODE + col;
        float* po_base = out      + (size_t)(t * 16 + mbase) * F_NODE + col;

#pragma unroll
        for (int nt = 0; nt < 8; ++nt) {
            v8f c = {};
#pragma unroll
            for (int ks = 0; ks < 4; ++ks) {
                v16h b = *(const v16h*)(wps_sw + (size_t)((nt * 4 + ks) * 32 + lane) * 16);
                c = __builtin_amdgcn_wmma_f32_16x16x32_f16(false, a[ks], false, b,
                                                           (short)0, c, false, false);
            }
            float bias = b_prop_src[nt * 16 + col];
            float* base = ps_base + nt * 16;
            if (full) {
#pragma unroll
                for (int v = 0; v < 8; ++v) base[v * F_NODE] = c[v] + bias;
            } else if (valid) {
#pragma unroll
                for (int v = 0; v < 8; ++v)
                    if (t * 16 + mbase + v < n_nodes) base[v * F_NODE] = c[v] + bias;
            }
        }

#pragma unroll
        for (int nt = 0; nt < 8; ++nt) {
            v8f c = {};
#pragma unroll
            for (int ks = 0; ks < 4; ++ks) {
                v16h b = *(const v16h*)(wpd_sw + (size_t)((nt * 4 + ks) * 32 + lane) * 16);
                c = __builtin_amdgcn_wmma_f32_16x16x32_f16(false, a[ks], false, b,
                                                           (short)0, c, false, false);
            }
            float bias = b_prop_dst[nt * 16 + col];
            float* base = po_base + nt * 16;
            if (full) {
#pragma unroll
                for (int v = 0; v < 8; ++v) base[v * F_NODE] = c[v] + bias;
            } else if (valid) {
#pragma unroll
                for (int v = 0; v < 8; ++v)
                    if (t * 16 + mbase + v < n_nodes) base[v * F_NODE] = c[v] + bias;
            }
        }
    }

    // ---- attention MLPs (src then dst): WMMA first layer, scalar second ----
    float* h = hid[wave];
    for (int which = 0; which < 2; ++which) {
        const _Float16* w0 = which ? wad0_sw : was0_sw;
        const float*    w1 = which ? w_att_dst1 : w_att_src1;
        float*          ao = which ? a_dst : a_src;

#pragma unroll
        for (int nt = 0; nt < 4; ++nt) {
            v8f c = {};
#pragma unroll
            for (int ks = 0; ks < 4; ++ks) {
                v16h b = *(const v16h*)(w0 + (size_t)((nt * 4 + ks) * 32 + lane) * 16);
                c = __builtin_amdgcn_wmma_f32_16x16x32_f16(false, a[ks], false, b,
                                                           (short)0, c, false, false);
            }
#pragma unroll
            for (int v = 0; v < 8; ++v)
                h[(mbase + v) * H_ATT + nt * 16 + col] = fmaxf(c[v], 0.0f);
        }
        __syncthreads();

        // second layer: 16 rows x 8 heads = 128 outputs, 4 per lane
        float* ao_base = ao + (size_t)t * 16 * HEADS;
#pragma unroll
        for (int u = 0; u < 4; ++u) {
            int idx = u * 32 + lane;
            int r  = idx >> 3;
            int hh = idx & 7;
            float s = 0.0f;
#pragma unroll
            for (int j = 0; j < H_ATT; ++j) s += h[r * H_ATT + j] * w1[j * HEADS + hh];
            if (full) {
                ao_base[idx] = s;
            } else if (valid && (t * 16 + r) < n_nodes) {
                ao_base[idx] = s;
            }
        }
        __syncthreads();
    }
}

// ---------------- edge MLP ----------------
__global__ void __launch_bounds__(256)
k_edge_mlp(const float* __restrict__ feat_edge,
           const float* __restrict__ w0,   // [8][64]
           const float* __restrict__ w1,   // [64][8]
           float* __restrict__ a_edge, int E) {
    __shared__ float sw0[8 * H_ATT];
    __shared__ float sw1[H_ATT * HEADS];
    for (int i = threadIdx.x; i < 8 * H_ATT; i += 256) { sw0[i] = w0[i]; sw1[i] = w1[i]; }
    __syncthreads();

    int e = blockIdx.x * 256 + threadIdx.x;
    if (e >= E) return;
    const float4 f0 = *(const float4*)(feat_edge + (size_t)e * 8);
    const float4 f1 = *(const float4*)(feat_edge + (size_t)e * 8 + 4);
    float f[8] = {f0.x, f0.y, f0.z, f0.w, f1.x, f1.y, f1.z, f1.w};
    float acc[8] = {0, 0, 0, 0, 0, 0, 0, 0};
    for (int j = 0; j < H_ATT; ++j) {
        float hsum = 0.0f;
#pragma unroll
        for (int k = 0; k < 8; ++k) hsum += f[k] * sw0[k * H_ATT + j];
        hsum = fmaxf(hsum, 0.0f);
#pragma unroll
        for (int o = 0; o < 8; ++o) acc[o] += hsum * sw1[j * HEADS + o];
    }
    float4* ae = (float4*)(a_edge + (size_t)e * 8);
    ae[0] = make_float4(acc[0], acc[1], acc[2], acc[3]);
    ae[1] = make_float4(acc[4], acc[5], acc[6], acc[7]);
}

// ---------------- attention logits + segment max ----------------
// att >= 0 after relu, so uint-bitcast atomicMax is exact float max (m init 0).
__global__ void __launch_bounds__(256)
k_att_max(const float* __restrict__ a_src, const float* __restrict__ a_dst,
          const float* __restrict__ a_edge, const int* __restrict__ src,
          const int* __restrict__ dst, float* __restrict__ att,
          unsigned int* __restrict__ m, int E) {
    int e = blockIdx.x * 256 + threadIdx.x;
    if (e >= E) return;
    int s = src[e], d = dst[e];
    const float4* as = (const float4*)(a_src + (size_t)s * HEADS);
    const float4* ad = (const float4*)(a_dst + (size_t)d * HEADS);
    const float4* ae = (const float4*)(a_edge + (size_t)e * HEADS);
    float v[8];
#pragma unroll
    for (int q = 0; q < 2; ++q) {
        float4 x = as[q], y = ad[q], z = ae[q];
        v[q * 4 + 0] = fmaxf(x.x + y.x + z.x, 0.0f);
        v[q * 4 + 1] = fmaxf(x.y + y.y + z.y, 0.0f);
        v[q * 4 + 2] = fmaxf(x.z + y.z + z.z, 0.0f);
        v[q * 4 + 3] = fmaxf(x.w + y.w + z.w, 0.0f);
    }
    float4* ao = (float4*)(att + (size_t)e * HEADS);
    ao[0] = make_float4(v[0], v[1], v[2], v[3]);
    ao[1] = make_float4(v[4], v[5], v[6], v[7]);
#pragma unroll
    for (int hh = 0; hh < HEADS; ++hh)
        atomicMax(&m[(size_t)d * HEADS + hh], __float_as_uint(v[hh]));
}

// ---------------- exp + segment sum ----------------
__global__ void __launch_bounds__(256)
k_exp_sum(float* __restrict__ att, const unsigned int* __restrict__ m,
          const int* __restrict__ dst, float* __restrict__ denom, int E) {
    int e = blockIdx.x * 256 + threadIdx.x;
    if (e >= E) return;
    int d = dst[e];
    const uint4* mp = (const uint4*)(m + (size_t)d * HEADS);
    float4* ap = (float4*)(att + (size_t)e * HEADS);
    float ex[8];
#pragma unroll
    for (int q = 0; q < 2; ++q) {
        uint4 mb = mp[q];
        float4 av = ap[q];
        ex[q * 4 + 0] = expf(av.x - __uint_as_float(mb.x));
        ex[q * 4 + 1] = expf(av.y - __uint_as_float(mb.y));
        ex[q * 4 + 2] = expf(av.z - __uint_as_float(mb.z));
        ex[q * 4 + 3] = expf(av.w - __uint_as_float(mb.w));
    }
    ap[0] = make_float4(ex[0], ex[1], ex[2], ex[3]);   // overwrite att with exp
    ap[1] = make_float4(ex[4], ex[5], ex[6], ex[7]);
#pragma unroll
    for (int hh = 0; hh < HEADS; ++hh)
        atomicAdd(&denom[(size_t)d * HEADS + hh], ex[hh]);
}

// ---------------- weighted message scatter ----------------
// One wave per edge; lane l handles floats [4l, 4l+3] of the 128-wide message.
__global__ void __launch_bounds__(256)
k_message(const float* __restrict__ prop_src, const float* __restrict__ ex,
          const float* __restrict__ denom, const int* __restrict__ src,
          const int* __restrict__ dst, float* __restrict__ out, int E) {
    int wave = threadIdx.x >> 5;
    int lane = threadIdx.x & 31;
    int e = blockIdx.x * 8 + wave;
    if (e >= E) return;
    int s = src[e], d = dst[e];
    int hh = lane >> 2;                                   // head = (4*lane)/16
    float alpha = ex[(size_t)e * HEADS + hh]
                / fmaxf(denom[(size_t)d * HEADS + hh], 1e-16f);
    const float4 p = *(const float4*)(prop_src + (size_t)s * F_NODE + lane * 4);
    float* o = out + (size_t)d * F_NODE + lane * 4;
    atomicAdd(o + 0, p.x * alpha);
    atomicAdd(o + 1, p.y * alpha);
    atomicAdd(o + 2, p.z * alpha);
    atomicAdd(o + 3, p.w * alpha);
}

// ---------------- host-side launcher ----------------

static inline size_t align256(size_t x) { return (x + 255) & ~(size_t)255; }

extern "C" void kernel_launch(void* const* d_in, const int* in_sizes, int n_in,
                              void* d_out, int out_size, void* d_ws, size_t ws_size,
                              hipStream_t stream) {
    const float* feat        = (const float*)d_in[0];
    const float* feat_edge   = (const float*)d_in[1];
    const int*   src         = (const int*)d_in[2];
    const int*   dst         = (const int*)d_in[3];
    const float* w_att_src0  = (const float*)d_in[4];
    const float* w_att_src1  = (const float*)d_in[5];
    const float* w_att_dst0  = (const float*)d_in[6];
    const float* w_att_dst1  = (const float*)d_in[7];
    const float* w_att_edge0 = (const float*)d_in[8];
    const float* w_att_edge1 = (const float*)d_in[9];
    const float* w_prop_src  = (const float*)d_in[10];
    const float* b_prop_src  = (const float*)d_in[11];
    const float* w_prop_dst  = (const float*)d_in[12];
    const float* b_prop_dst  = (const float*)d_in[13];

    const int N = in_sizes[0] / F_NODE;
    const int E = in_sizes[2];

    // carve workspace
    char* ws = (char*)d_ws;
    size_t off = 0;
    auto carve = [&](size_t bytes) -> void* {
        void* p = ws + off;
        off = align256(off + bytes);
        return p;
    };
    _Float16* feat16   = (_Float16*)carve((size_t)N * F_NODE * sizeof(_Float16));
    _Float16* wps_sw   = (_Float16*)carve(8 * 4 * 32 * 16 * sizeof(_Float16));
    _Float16* wpd_sw   = (_Float16*)carve(8 * 4 * 32 * 16 * sizeof(_Float16));
    _Float16* was0_sw  = (_Float16*)carve(4 * 4 * 32 * 16 * sizeof(_Float16));
    _Float16* wad0_sw  = (_Float16*)carve(4 * 4 * 32 * 16 * sizeof(_Float16));
    float*    prop_src = (float*)carve((size_t)N * F_NODE * sizeof(float));
    float*    a_src    = (float*)carve((size_t)N * HEADS * sizeof(float));
    float*    a_dst    = (float*)carve((size_t)N * HEADS * sizeof(float));
    float*    a_edge   = (float*)carve((size_t)E * HEADS * sizeof(float));
    float*    att      = (float*)carve((size_t)E * HEADS * sizeof(float));
    unsigned* m_bits   = (unsigned*)carve((size_t)N * HEADS * sizeof(unsigned));
    float*    denom    = (float*)carve((size_t)N * HEADS * sizeof(float));
    (void)ws_size; (void)n_in; (void)out_size;

    float* out = (float*)d_out;

    // 1) precision/layout prep
    k_f32_to_f16<<<((size_t)N * F_NODE + 255) / 256, 256, 0, stream>>>(feat, feat16, N * F_NODE);
    k_swizzle_b<<<(16384 + 255) / 256, 256, 0, stream>>>(w_prop_src, wps_sw, 128, 128, 4, 16384);
    k_swizzle_b<<<(16384 + 255) / 256, 256, 0, stream>>>(w_prop_dst, wpd_sw, 128, 128, 4, 16384);
    k_swizzle_b<<<(8192 + 255) / 256, 256, 0, stream>>>(w_att_src0, was0_sw, 128, 64, 4, 8192);
    k_swizzle_b<<<(8192 + 255) / 256, 256, 0, stream>>>(w_att_dst0, wad0_sw, 128, 64, 4, 8192);

    // 2) fused node GEMMs (WMMA): prop_src, out(=prop_dst), a_src, a_dst
    int n_tiles = (N + 15) / 16;
    k_node_fused<<<(n_tiles + 7) / 8, 256, 0, stream>>>(
        feat16, wps_sw, wpd_sw, was0_sw, wad0_sw,
        w_att_src1, w_att_dst1, b_prop_src, b_prop_dst,
        prop_src, a_src, a_dst, out, n_tiles, N);

    // 3) zero softmax accumulators (deterministic per-call)
    k_zero_u32<<<(N * HEADS + 255) / 256, 256, 0, stream>>>(m_bits, N * HEADS);
    k_zero_u32<<<(N * HEADS + 255) / 256, 256, 0, stream>>>((unsigned*)denom, N * HEADS);

    // 4) edge MLP
    k_edge_mlp<<<((size_t)E + 255) / 256, 256, 0, stream>>>(feat_edge, w_att_edge0, w_att_edge1,
                                                            a_edge, E);

    // 5) attention logits + segment max
    k_att_max<<<((size_t)E + 255) / 256, 256, 0, stream>>>(a_src, a_dst, a_edge, src, dst,
                                                           att, m_bits, E);

    // 6) exp + segment sum
    k_exp_sum<<<((size_t)E + 255) / 256, 256, 0, stream>>>(att, m_bits, dst, denom, E);

    // 7) weighted scatter of messages into out (residual already present)
    k_message<<<((size_t)E + 7) / 8, 256, 0, stream>>>(prop_src, att, denom, src, dst, out, E);
}
